// EGraphSAGELayer_56057913147664
// MI455X (gfx1250) — matrix operand order repaired and checked by reading, not compiled, not run on hardware
//
#include <hip/hip_runtime.h>

#define N_NODES  100000
#define N_EDGES  1600000
#define NODE_IN  64
#define EDGE_DIM 32
#define NODE_OUT 64

typedef __attribute__((ext_vector_type(16))) _Float16 v16h;
typedef __attribute__((ext_vector_type(8)))  float    v8f;

// ---- A-fragment loader: 16x32 f16 A tile, one row per lane-pair ----
// Lane L = 16h + m holds row m; VGPR v (packed pair) holds
//   K = 8h + 2v + j          for v in 0..3
//   K = 16 + 8h + 2(v-4) + j for v in 4..7
__device__ __forceinline__ v16h load_a_f32(const float* __restrict__ row, int h) {
  v16h a;
#pragma unroll
  for (int v = 0; v < 8; ++v) {
    int kk = (v < 4) ? (8 * h + 2 * v) : (16 + 8 * h + 2 * (v - 4));
    a[2 * v]     = (_Float16)row[kk];
    a[2 * v + 1] = (_Float16)row[kk + 1];
  }
  return a;
}

__device__ __forceinline__ v16h load_a_f32_scaled(const float* __restrict__ row, int h, float sc) {
  v16h a;
#pragma unroll
  for (int v = 0; v < 8; ++v) {
    int kk = (v < 4) ? (8 * h + 2 * v) : (16 + 8 * h + 2 * (v - 4));
    a[2 * v]     = (_Float16)(row[kk] * sc);
    a[2 * v + 1] = (_Float16)(row[kk + 1] * sc);
  }
  return a;
}

// ---- B-fragment loader: 32x16 f16 B tile from row-major f32 W[K][64] ----
// Lane L = 16h + n holds column (16*ntile + n); VGPR v holds K = kbase + 8h + 2v + j (v<4), kbase + 16 + 8h + 2(v-4) + j (v>=4)
__device__ __forceinline__ v16h load_b_frag(const float* __restrict__ W, int kbase,
                                            int ntile, int h, int n) {
  v16h b;
#pragma unroll
  for (int v = 0; v < 8; ++v) {
    int k0 = kbase + ((v < 4) ? (8 * h + 2 * v) : (16 + 8 * h + 2 * (v - 4)));
    b[2 * v]     = (_Float16)W[(size_t)k0 * 64 + 16 * ntile + n];
    b[2 * v + 1] = (_Float16)W[(size_t)(k0 + 1) * 64 + 16 * ntile + n];
  }
  return b;
}

// ---- Kernel 0: zero workspace accumulators ----
__global__ void __launch_bounds__(256) zero_ws_kernel(float* __restrict__ p, int n) {
  int i = blockIdx.x * blockDim.x + threadIdx.x;
  int stride = gridDim.x * blockDim.x;
  for (; i < n; i += stride) p[i] = 0.0f;
}

// ---- Kernel 1: edge messages (GEMM) + scatter-add into s / cnt ----
__global__ void __launch_bounds__(256)
edge_msg_kernel(const float* __restrict__ nfeats,
                const float* __restrict__ efeats,
                const int*   __restrict__ src,
                const int*   __restrict__ dst,
                const float* __restrict__ Wm,   // [96][64]
                const float* __restrict__ bm,   // [64]
                float* __restrict__ s_acc,      // [N][64]
                float* __restrict__ cnt)        // [N]
{
  const int lane = threadIdx.x & 31;
  const int h = lane >> 4;
  const int m = lane & 15;   // row within tile (edge)
  const int n = m;           // column within N-tile
  const int wavesPerBlock = blockDim.x >> 5;
  const int wave  = blockIdx.x * wavesPerBlock + (threadIdx.x >> 5);
  const int nWave = gridDim.x * wavesPerBlock;

  // Preload all B fragments of W_msg: 3 K-steps x 4 N-tiles (96 VGPRs, hits L2/L1)
  v16h B[3][4];
#pragma unroll
  for (int ks = 0; ks < 3; ++ks)
#pragma unroll
    for (int nt = 0; nt < 4; ++nt)
      B[ks][nt] = load_b_frag(Wm, 32 * ks, nt, h, n);

  float bias[4];
#pragma unroll
  for (int nt = 0; nt < 4; ++nt) bias[nt] = bm[16 * nt + n];

  const int nTiles = N_EDGES / 16;  // 100000, exact
  for (int t = wave; t < nTiles; t += nWave) {
    const int e    = t * 16 + m;
    const int sIdx = src[e];
    const int dIdx = dst[e];

    // A fragments: K 0..31 / 32..63 from gathered node row (L2-resident),
    // K 64..95 from streaming edge row.
    v16h a0 = load_a_f32(nfeats + (size_t)sIdx * NODE_IN, h);
    v16h a1 = load_a_f32(nfeats + (size_t)sIdx * NODE_IN + 32, h);
    v16h a2 = load_a_f32(efeats + (size_t)e * EDGE_DIM, h);

#pragma unroll
    for (int nt = 0; nt < 4; ++nt) {
      v8f c = {0.f, 0.f, 0.f, 0.f, 0.f, 0.f, 0.f, 0.f};
      c = __builtin_amdgcn_wmma_f32_16x16x32_f16(false, a0, false, B[0][nt], (short)0, c, false, false);
      c = __builtin_amdgcn_wmma_f32_16x16x32_f16(false, a1, false, B[1][nt], (short)0, c, false, false);
      c = __builtin_amdgcn_wmma_f32_16x16x32_f16(false, a2, false, B[2][nt], (short)0, c, false, false);
      // D layout: VGPR r holds element (M = r + 8h, N = 16*nt + n).
#pragma unroll
      for (int r = 0; r < 8; ++r) {
        int M  = r + 8 * h;
        int dd = __shfl(dIdx, M, 32);   // dst of row M lives in lane M
        atomicAdd(&s_acc[(size_t)dd * NODE_OUT + 16 * nt + n], c[r] + bias[nt]);
      }
    }
    if (lane < 16) atomicAdd(&cnt[dIdx], 1.0f);
  }
}

// ---- Kernel 2: node apply GEMM (concat[nfeats, s/cnt] @ W_apply + b, ReLU) ----
__global__ void __launch_bounds__(256)
node_apply_kernel(const float* __restrict__ nfeats,
                  const float* __restrict__ Wa,  // [128][64]
                  const float* __restrict__ ba,  // [64]
                  const float* __restrict__ s_acc,
                  const float* __restrict__ cnt,
                  float* __restrict__ out)       // [N][64]
{
  const int lane = threadIdx.x & 31;
  const int h = lane >> 4;
  const int m = lane & 15;
  const int n = m;
  const int wavesPerBlock = blockDim.x >> 5;
  const int wave  = blockIdx.x * wavesPerBlock + (threadIdx.x >> 5);
  const int nWave = gridDim.x * wavesPerBlock;

  v16h B[4][4];  // 4 K-steps x 4 N-tiles of W_apply
#pragma unroll
  for (int ks = 0; ks < 4; ++ks)
#pragma unroll
    for (int nt = 0; nt < 4; ++nt)
      B[ks][nt] = load_b_frag(Wa, 32 * ks, nt, h, n);

  float bias[4];
#pragma unroll
  for (int nt = 0; nt < 4; ++nt) bias[nt] = ba[16 * nt + n];

  const int nTiles = N_NODES / 16;  // 6250, exact
  for (int t = wave; t < nTiles; t += nWave) {
    const int node = t * 16 + m;
    const float inv = 1.0f / fmaxf(cnt[node], 1.0f);  // mean; isolated nodes -> 0

    v16h a0 = load_a_f32(nfeats + (size_t)node * NODE_IN, h);
    v16h a1 = load_a_f32(nfeats + (size_t)node * NODE_IN + 32, h);
    v16h a2 = load_a_f32_scaled(s_acc + (size_t)node * NODE_OUT, h, inv);
    v16h a3 = load_a_f32_scaled(s_acc + (size_t)node * NODE_OUT + 32, h, inv);

#pragma unroll
    for (int nt = 0; nt < 4; ++nt) {
      v8f c = {0.f, 0.f, 0.f, 0.f, 0.f, 0.f, 0.f, 0.f};
      c = __builtin_amdgcn_wmma_f32_16x16x32_f16(false, a0, false, B[0][nt], (short)0, c, false, false);
      c = __builtin_amdgcn_wmma_f32_16x16x32_f16(false, a1, false, B[1][nt], (short)0, c, false, false);
      c = __builtin_amdgcn_wmma_f32_16x16x32_f16(false, a2, false, B[2][nt], (short)0, c, false, false);
      c = __builtin_amdgcn_wmma_f32_16x16x32_f16(false, a3, false, B[3][nt], (short)0, c, false, false);
#pragma unroll
      for (int r = 0; r < 8; ++r) {
        int M = r + 8 * h;
        int nodeM = t * 16 + M;
        out[(size_t)nodeM * NODE_OUT + 16 * nt + n] = fmaxf(c[r] + bias[nt], 0.0f);
      }
    }
  }
}

extern "C" void kernel_launch(void* const* d_in, const int* in_sizes, int n_in,
                              void* d_out, int out_size, void* d_ws, size_t ws_size,
                              hipStream_t stream) {
  const float* nfeats = (const float*)d_in[0];
  const float* efeats = (const float*)d_in[1];
  const int*   src    = (const int*)d_in[2];
  const int*   dst    = (const int*)d_in[3];
  const float* Wm     = (const float*)d_in[4];
  const float* bm     = (const float*)d_in[5];
  const float* Wa     = (const float*)d_in[6];
  const float* ba     = (const float*)d_in[7];
  float* out   = (float*)d_out;
  float* s_acc = (float*)d_ws;                       // [N_NODES][64]
  float* cnt   = s_acc + (size_t)N_NODES * NODE_OUT; // [N_NODES]

  const int wsElems = N_NODES * NODE_OUT + N_NODES;
  zero_ws_kernel<<<2048, 256, 0, stream>>>(s_acc, wsElems);

  // 2048 blocks x 8 waves = 16384 waves over 100000 edge tiles (~6 tiles/wave,
  // amortizes the in-register W_msg fragments).
  edge_msg_kernel<<<2048, 256, 0, stream>>>(nfeats, efeats, src, dst, Wm, bm, s_acc, cnt);

  // 782 blocks x 8 waves covers the 6250 node tiles.
  node_apply_kernel<<<782, 256, 0, stream>>>(nfeats, Wa, ba, s_acc, cnt, out);
}
